// GCLayer_54296976556722
// MI455X (gfx1250) — matrix operand order repaired and checked by reading, not compiled, not run on hardware
//
#include <hip/hip_runtime.h>

// ---------------------------------------------------------------- types
typedef __attribute__((ext_vector_type(16))) __bf16 v16bf;
typedef __attribute__((ext_vector_type(8)))  float  v8f;

union ABits { uint4 u[2]; v16bf v; };

#define WMMA_BF16(A, B, C) \
    __builtin_amdgcn_wmma_f32_16x16x32_bf16(false, (A), false, (B), (short)0, (C), false, false)

__device__ __forceinline__ float dev_silu(float f) {
    return f / (1.0f + __expf(-f));
}

// Load a 16x32 bf16 A fragment from a row-major bf16 row (K-chunk base).
// Lane l<16: M=l, elems 0..7 = K 0..7, elems 8..15 = K 16..23
// Lane l>=16: M=l-16, elems 0..7 = K 8..15, elems 8..15 = K 24..31
__device__ __forceinline__ v16bf load_a_bf16(const __bf16* rowp, int hi) {
    ABits ab;
    ab.u[0] = *(const uint4*)(rowp + hi * 8);
    ab.u[1] = *(const uint4*)(rowp + 16 + hi * 8);
    return ab.v;
}

// Same A layout, fp32 source (optionally applying silu first).
__device__ __forceinline__ v16bf load_a_f32(const float* rowp, int hi, int do_silu) {
    v16bf r;
#pragma unroll
    for (int i = 0; i < 8; ++i) {
        float f0 = rowp[hi * 8 + i];
        float f1 = rowp[16 + hi * 8 + i];
        if (do_silu) { f0 = dev_silu(f0); f1 = dev_silu(f1); }
        r[i]     = (__bf16)f0;
        r[8 + i] = (__bf16)f1;
    }
    return r;
}

// B fragment: packed as [tile][lane][16 bf16] contiguous (LDS or global ptr).
__device__ __forceinline__ v16bf load_b(const __bf16* frag, int lane) {
    ABits ab;
    const __bf16* p = frag + lane * 16;
    ab.u[0] = ((const uint4*)p)[0];
    ab.u[1] = ((const uint4*)p)[1];
    return ab.v;
}

// Reload an A fragment from an LDS-staged row-major 16x128 bf16 tile.
__device__ __forceinline__ v16bf load_a_lds(const __bf16* tile, int kc, int lane) {
    int m = lane & 15, hi = lane >> 4;
    const __bf16* p = tile + m * 128 + kc * 32;
    ABits ab;
    ab.u[0] = *(const uint4*)(p + hi * 8);
    ab.u[1] = *(const uint4*)(p + 16 + hi * 8);
    return ab.v;
}

// K-chain with software-pipelined B prefetch (one M-tile).
template <int KC>
__device__ __forceinline__ v8f gemm_chain(const v16bf* A, const __bf16* Bbase,
                                          int strideElems, int lane, v8f acc) {
    v16bf b = load_b(Bbase, lane);
#pragma unroll
    for (int kc = 0; kc < KC; ++kc) {
        v16bf bn = b;
        if (kc + 1 < KC) bn = load_b(Bbase + (size_t)(kc + 1) * strideElems, lane);
        acc = WMMA_BF16(A[kc], b, acc);
        b = bn;
    }
    return acc;
}

// Dual-M K-chain: each B fragment feeds TWO WMMAs (two M-tiles), halving
// B-side LDS bandwidth per WMMA. Accumulators are disjoint -> no WMMA hazard.
template <int KC>
__device__ __forceinline__ void gemm_chain2(const v16bf* A0, const v16bf* A1,
                                            const __bf16* Bbase, int strideElems,
                                            int lane, v8f& acc0, v8f& acc1) {
    v16bf b = load_b(Bbase, lane);
#pragma unroll
    for (int kc = 0; kc < KC; ++kc) {
        v16bf bn = b;
        if (kc + 1 < KC) bn = load_b(Bbase + (size_t)(kc + 1) * strideElems, lane);
        acc0 = WMMA_BF16(A0[kc], b, acc0);
        acc1 = WMMA_BF16(A1[kc], b, acc1);
        b = bn;
    }
}

// Block-cooperative 16B-chunk copy into LDS.
__device__ __forceinline__ void stage_lds(void* dst, const void* src, int nbytes,
                                          int tid, int nthr) {
    uint4* d = (uint4*)dst;
    const uint4* s = (const uint4*)src;
    int n16 = nbytes >> 4;
    for (int i = tid; i < n16; i += nthr) d[i] = s[i];
}

// ---------------------------------------------------------------- utility kernels

__global__ __launch_bounds__(256) void zero_f4_kernel(float4* p, long n4) {
    long i = (long)blockIdx.x * blockDim.x + threadIdx.x;
    if (i < n4) p[i] = make_float4(0.f, 0.f, 0.f, 0.f);
}

// Pack W[K x N] (row-major fp32) into bf16 WMMA B fragments.
__global__ __launch_bounds__(256) void pack_b_kernel(const float* __restrict__ W,
                                                     __bf16* __restrict__ dst,
                                                     int K, int N) {
    int t    = blockIdx.x * blockDim.x + threadIdx.x;
    int lane = t & 31;
    int tile = t >> 5;
    int ntn  = N >> 4;
    int nTiles = (K >> 5) * ntn;
    if (tile >= nTiles) return;
    int kt = tile / ntn, nt = tile % ntn;
    int n     = nt * 16 + (lane & 15);
    int kbase = kt * 32 + (lane >> 4) * 8;
    __bf16* out = dst + (size_t)tile * 512 + lane * 16;
#pragma unroll
    for (int e = 0; e < 16; ++e)
        out[e] = (__bf16)W[(size_t)(kbase + e + (e & 8)) * N + n];
}

// Pack W_att[256] as B fragments of a 256x16 matrix with column 0 = W_att.
__global__ void pack_att_kernel(const float* __restrict__ W_att, __bf16* __restrict__ dst) {
    int t = blockIdx.x * blockDim.x + threadIdx.x;   // 8 tiles * 32 lanes
    int lane = t & 31, kt = t >> 5;
    if (kt >= 8) return;
    int n     = lane & 15;
    int kbase = kt * 32 + (lane >> 4) * 8;
    __bf16* out = dst + (size_t)kt * 512 + lane * 16;
#pragma unroll
    for (int e = 0; e < 16; ++e)
        out[e] = (__bf16)((n == 0) ? W_att[kbase + e + (e & 8)] : 0.0f);
}

__global__ void bias_comb_kernel(const float* a, const float* b, float* c, int n) {
    int i = blockIdx.x * blockDim.x + threadIdx.x;
    if (i < n) c[i] = a[i] + b[i];
}

// ---------------------------------------------------------------- node projection
// h = (x @ W_lin + silu(temb) @ Wt + (b_lin+bt)) @ W_lin1
__global__ __launch_bounds__(256) void node_proj_kernel(
    const float* __restrict__ x, const float* __restrict__ temb,
    const __bf16* __restrict__ Blin, const __bf16* __restrict__ Bt,
    const __bf16* __restrict__ Blin1, const float* __restrict__ bc,
    float* __restrict__ h32, __bf16* __restrict__ hbf, int Nn) {
    __shared__ __align__(16) __bf16 sBlin[32 * 512];
    __shared__ __align__(16) __bf16 sBt[32 * 512];
    __shared__ __align__(16) __bf16 sBlin1[32 * 512];
    __shared__ __align__(16) __bf16 sT1[8][16 * 128];

    int tid = threadIdx.x;
    stage_lds(sBlin,  Blin,  32 * 512 * 2, tid, 256);
    stage_lds(sBt,    Bt,    32 * 512 * 2, tid, 256);
    stage_lds(sBlin1, Blin1, 32 * 512 * 2, tid, 256);
    __syncthreads();

    int lane = tid & 31;
    int wv   = tid >> 5;
    int m = lane & 15, hi = lane >> 4;
    int nTiles = Nn / 16;

    float bcv[8];
#pragma unroll
    for (int n = 0; n < 8; ++n) bcv[n] = bc[n * 16 + m];

    __bf16* t1 = sT1[wv];
    for (int tile = blockIdx.x * 8 + wv; tile < nTiles; tile += gridDim.x * 8) {
        int rowbase = tile * 16;
        const float* xrow = x    + (size_t)(rowbase + m) * 128;
        const float* trow = temb + (size_t)(rowbase + m) * 128;
        v16bf Ax[4], At[4];
#pragma unroll
        for (int kc = 0; kc < 4; ++kc) {
            Ax[kc] = load_a_f32(xrow + kc * 32, hi, 0);
            At[kc] = load_a_f32(trow + kc * 32, hi, 1);
        }

#pragma unroll
        for (int n = 0; n < 8; ++n) {
            v8f acc = {};
            acc = gemm_chain<4>(Ax, sBlin + n * 512, 4096, lane, acc);
            acc = gemm_chain<4>(At, sBt   + n * 512, 4096, lane, acc);
#pragma unroll
            for (int v = 0; v < 8; ++v)
                t1[(v + 8 * hi) * 128 + n * 16 + m] = (__bf16)(acc[v] + bcv[n]);
        }
        asm volatile("s_wait_dscnt 0" ::: "memory");

        v16bf A2[4];
#pragma unroll
        for (int kc = 0; kc < 4; ++kc) A2[kc] = load_a_lds(t1, kc, lane);

#pragma unroll
        for (int n = 0; n < 8; ++n) {
            v8f acc = {};
            acc = gemm_chain<4>(A2, sBlin1 + n * 512, 4096, lane, acc);
#pragma unroll
            for (int v = 0; v < 8; ++v) {
                int r = rowbase + v + 8 * hi;
                int c = n * 16 + m;
                h32[(size_t)r * 128 + c] = acc[v];
                hbf[(size_t)r * 128 + c] = (__bf16)acc[v];
            }
        }
    }
}

// ---------------------------------------------------------------- edge kernel
// 32 edges (two 16-row M-tiles) per wave; every B fragment feeds 2 WMMAs.
// att = sigmoid(cat @ W_att + b_att) * edge_mask      (via WMMA, col 0)
// msg = silu(silu(cat @ We1 + be1) @ We2 + be2)
// agg[row] += msg * att                               (fp32 atomics)
__global__ __launch_bounds__(192) void edge_kernel(
    const __bf16* __restrict__ hbf,
    const int* __restrict__ row_idx, const int* __restrict__ col_idx,
    const float* __restrict__ b_att, const float* __restrict__ edge_mask,
    const __bf16* __restrict__ Be1, const float* __restrict__ be1,
    const __bf16* __restrict__ Be2, const float* __restrict__ be2,
    const __bf16* __restrict__ Batt,
    float* __restrict__ agg, int Ecnt) {
    __shared__ __align__(16) __bf16 sBe1[64 * 512];    // 64 KB
    __shared__ __align__(16) __bf16 sBe2[32 * 512];    // 32 KB
    __shared__ __align__(16) __bf16 sBatt[8 * 512];    //  8 KB
    __shared__ __align__(16) __bf16 sT1[6][32 * 128];  // 48 KB
    __shared__ __align__(16) float  att_s[6][32];

    int tid = threadIdx.x;
    stage_lds(sBe1,  Be1,  64 * 512 * 2, tid, 192);
    stage_lds(sBe2,  Be2,  32 * 512 * 2, tid, 192);
    stage_lds(sBatt, Batt,  8 * 512 * 2, tid, 192);
    __syncthreads();

    int lane = tid & 31;
    int wv   = tid >> 5;             // 0..5
    int m = lane & 15, hi = lane >> 4;
    int nTiles = Ecnt / 32;

    float be1v[8], be2v[8];
#pragma unroll
    for (int n = 0; n < 8; ++n) { be1v[n] = be1[n * 16 + m]; be2v[n] = be2[n * 16 + m]; }
    const float batt = b_att[0];

    __bf16* t1 = sT1[wv];
    for (int tp = blockIdx.x * 6 + wv; tp < nTiles; tp += gridDim.x * 6) {
        int base = tp * 32;

        v16bf A0[8], A1[8];
        {
            int e0 = base + m, e1 = base + 16 + m;
            const __bf16* r0 = hbf + (size_t)row_idx[e0] * 128;
            const __bf16* c0 = hbf + (size_t)col_idx[e0] * 128;
            const __bf16* r1 = hbf + (size_t)row_idx[e1] * 128;
            const __bf16* c1 = hbf + (size_t)col_idx[e1] * 128;
#pragma unroll
            for (int kc = 0; kc < 4; ++kc) {
                A0[kc]     = load_a_bf16(r0 + kc * 32, hi);
                A0[4 + kc] = load_a_bf16(c0 + kc * 32, hi);
                A1[kc]     = load_a_bf16(r1 + kc * 32, hi);
                A1[4 + kc] = load_a_bf16(c1 + kc * 32, hi);
            }
        }

        // attention logits via WMMA: column 0 of D holds cat . W_att
        {
            v8f a0 = {}, a1 = {};
            gemm_chain2<8>(A0, A1, sBatt, 512, lane, a0, a1);
            if (m == 0) {   // lanes 0 and 16 own column 0, rows hi*8..hi*8+7
#pragma unroll
                for (int v = 0; v < 8; ++v) {
                    int rl = v + 8 * hi;
                    float f0 = 1.0f / (1.0f + __expf(-(a0[v] + batt)));
                    float f1 = 1.0f / (1.0f + __expf(-(a1[v] + batt)));
                    att_s[wv][rl]      = f0 * edge_mask[base + rl];
                    att_s[wv][16 + rl] = f1 * edge_mask[base + 16 + rl];
                }
            }
        }

        // GEMM1: t1 = silu(cat @ We1 + be1) -> LDS (32 rows)
#pragma unroll
        for (int n = 0; n < 8; ++n) {
            v8f a0 = {}, a1 = {};
            gemm_chain2<8>(A0, A1, sBe1 + n * 512, 4096, lane, a0, a1);
#pragma unroll
            for (int v = 0; v < 8; ++v) {
                t1[(v + 8 * hi) * 128 + n * 16 + m]        = (__bf16)dev_silu(a0[v] + be1v[n]);
                t1[(16 + v + 8 * hi) * 128 + n * 16 + m]   = (__bf16)dev_silu(a1[v] + be1v[n]);
            }
        }
        asm volatile("s_wait_dscnt 0" ::: "memory");

        v16bf A20[4], A21[4];
#pragma unroll
        for (int kc = 0; kc < 4; ++kc) {
            A20[kc] = load_a_lds(t1, kc, lane);
            A21[kc] = load_a_lds(t1 + 16 * 128, kc, lane);
        }

        float attv0[8], attv1[8];
        {
            const float4* ap = (const float4*)&att_s[wv][hi * 8];
            float4 a0 = ap[0], a1 = ap[1];
            attv0[0] = a0.x; attv0[1] = a0.y; attv0[2] = a0.z; attv0[3] = a0.w;
            attv0[4] = a1.x; attv0[5] = a1.y; attv0[6] = a1.z; attv0[7] = a1.w;
            const float4* bp = (const float4*)&att_s[wv][16 + hi * 8];
            float4 b0 = bp[0], b1 = bp[1];
            attv1[0] = b0.x; attv1[1] = b0.y; attv1[2] = b0.z; attv1[3] = b0.w;
            attv1[4] = b1.x; attv1[5] = b1.y; attv1[6] = b1.z; attv1[7] = b1.w;
        }
        int nid0[8], nid1[8];
        {
            const int4* ip = (const int4*)&row_idx[base + hi * 8];
            int4 i0 = ip[0], i1 = ip[1];
            nid0[0] = i0.x; nid0[1] = i0.y; nid0[2] = i0.z; nid0[3] = i0.w;
            nid0[4] = i1.x; nid0[5] = i1.y; nid0[6] = i1.z; nid0[7] = i1.w;
            const int4* jp = (const int4*)&row_idx[base + 16 + hi * 8];
            int4 j0 = jp[0], j1 = jp[1];
            nid1[0] = j0.x; nid1[1] = j0.y; nid1[2] = j0.z; nid1[3] = j0.w;
            nid1[4] = j1.x; nid1[5] = j1.y; nid1[6] = j1.z; nid1[7] = j1.w;
        }

        // GEMM2: msg = silu(t1 @ We2 + be2) * att ; scatter-add
#pragma unroll
        for (int n = 0; n < 8; ++n) {
            v8f a0 = {}, a1 = {};
            gemm_chain2<4>(A20, A21, sBe2 + n * 512, 4096, lane, a0, a1);
#pragma unroll
            for (int v = 0; v < 8; ++v) {
                float f0 = dev_silu(a0[v] + be2v[n]) * attv0[v];
                float f1 = dev_silu(a1[v] + be2v[n]) * attv1[v];
                atomicAdd(&agg[(size_t)nid0[v] * 128 + n * 16 + m], f0);
                atomicAdd(&agg[(size_t)nid1[v] * 128 + n * 16 + m], f1);
            }
        }
    }
}

// ---------------------------------------------------------------- node MLP
// out = (h + silu([h, agg] @ Wn1 + bn1) @ Wn2 + bn2) * node_mask
__global__ __launch_bounds__(256) void node_mlp_kernel(
    const float* __restrict__ h32, const __bf16* __restrict__ hbf,
    const float* __restrict__ agg,
    const __bf16* __restrict__ Bn1, const float* __restrict__ bn1,
    const __bf16* __restrict__ Bn2, const float* __restrict__ bn2,
    const float* __restrict__ node_mask, float* __restrict__ out, int Nn) {
    __shared__ __align__(16) __bf16 sBn1[64 * 512];
    __shared__ __align__(16) __bf16 sBn2[32 * 512];
    __shared__ __align__(16) __bf16 sT1[8][16 * 128];

    int tid = threadIdx.x;
    stage_lds(sBn1, Bn1, 64 * 512 * 2, tid, 256);
    stage_lds(sBn2, Bn2, 32 * 512 * 2, tid, 256);
    __syncthreads();

    int lane = tid & 31;
    int wv   = tid >> 5;
    int m = lane & 15, hi = lane >> 4;
    int nTiles = Nn / 16;

    float bn1v[8], bn2v[8];
#pragma unroll
    for (int n = 0; n < 8; ++n) { bn1v[n] = bn1[n * 16 + m]; bn2v[n] = bn2[n * 16 + m]; }

    __bf16* t1 = sT1[wv];
    for (int tile = blockIdx.x * 8 + wv; tile < nTiles; tile += gridDim.x * 8) {
        int rowbase = tile * 16;
        const __bf16* hrow = hbf + (size_t)(rowbase + m) * 128;
        const float*  arow = agg + (size_t)(rowbase + m) * 128;
        v16bf A[8];
#pragma unroll
        for (int kc = 0; kc < 4; ++kc) {
            A[kc]     = load_a_bf16(hrow + kc * 32, hi);
            A[4 + kc] = load_a_f32(arow + kc * 32, hi, 0);
        }

#pragma unroll
        for (int n = 0; n < 8; ++n) {
            v8f acc = {};
            acc = gemm_chain<8>(A, sBn1 + n * 512, 4096, lane, acc);
#pragma unroll
            for (int v = 0; v < 8; ++v)
                t1[(v + 8 * hi) * 128 + n * 16 + m] = (__bf16)dev_silu(acc[v] + bn1v[n]);
        }
        asm volatile("s_wait_dscnt 0" ::: "memory");

        v16bf A2[4];
#pragma unroll
        for (int kc = 0; kc < 4; ++kc) A2[kc] = load_a_lds(t1, kc, lane);

        float nm[8];
        {
            const float* mp = node_mask + rowbase + hi * 8;
#pragma unroll
            for (int v = 0; v < 8; ++v) nm[v] = mp[v];
        }

#pragma unroll
        for (int n = 0; n < 8; ++n) {
            v8f acc = {};
            acc = gemm_chain<4>(A2, sBn2 + n * 512, 4096, lane, acc);
#pragma unroll
            for (int v = 0; v < 8; ++v) {
                int r = rowbase + v + 8 * hi;
                int c = n * 16 + m;
                float f = h32[(size_t)r * 128 + c] + acc[v] + bn2v[n];
                out[(size_t)r * 128 + c] = f * nm[v];
            }
        }
    }
}

// ---------------------------------------------------------------- host
extern "C" void kernel_launch(void* const* d_in, const int* in_sizes, int n_in,
                              void* d_out, int out_size, void* d_ws, size_t ws_size,
                              hipStream_t stream) {
    const float* x         = (const float*)d_in[0];
    const int*   edges     = (const int*)d_in[1];
    const float* node_mask = (const float*)d_in[2];
    const float* edge_mask = (const float*)d_in[3];
    const float* temb      = (const float*)d_in[4];
    const float* W_lin     = (const float*)d_in[5];
    const float* b_lin     = (const float*)d_in[6];
    const float* W_lin1    = (const float*)d_in[7];
    const float* Wt        = (const float*)d_in[8];
    const float* bt        = (const float*)d_in[9];
    const float* W_att     = (const float*)d_in[10];
    const float* b_att     = (const float*)d_in[11];
    const float* We1       = (const float*)d_in[12];
    const float* be1       = (const float*)d_in[13];
    const float* We2       = (const float*)d_in[14];
    const float* be2       = (const float*)d_in[15];
    const float* Wn1       = (const float*)d_in[16];
    const float* bn1       = (const float*)d_in[17];
    const float* Wn2       = (const float*)d_in[18];
    const float* bn2       = (const float*)d_in[19];

    const int Nn   = in_sizes[0] / 128;   // 50000
    const int Ecnt = in_sizes[1] / 2;     // 800000
    const int* row_idx = edges;
    const int* col_idx = edges + Ecnt;

    char* p = (char*)d_ws;
    float*  h32  = (float*)p;   p += (size_t)Nn * 128 * sizeof(float);
    __bf16* hbf  = (__bf16*)p;  p += (size_t)Nn * 128 * sizeof(__bf16);
    float*  agg  = (float*)p;   p += (size_t)Nn * 128 * sizeof(float);
    __bf16* Blin  = (__bf16*)p; p += (size_t)32 * 512 * 2;  // 128x128
    __bf16* Bt    = (__bf16*)p; p += (size_t)32 * 512 * 2;
    __bf16* Blin1 = (__bf16*)p; p += (size_t)32 * 512 * 2;
    __bf16* Be1   = (__bf16*)p; p += (size_t)64 * 512 * 2;  // 256x128
    __bf16* Be2   = (__bf16*)p; p += (size_t)32 * 512 * 2;
    __bf16* Bn1   = (__bf16*)p; p += (size_t)64 * 512 * 2;
    __bf16* Bn2   = (__bf16*)p; p += (size_t)32 * 512 * 2;
    __bf16* Batt  = (__bf16*)p; p += (size_t)8  * 512 * 2;  // 256x16 (col 0 = W_att)
    float*  bc    = (float*)p;  p += 128 * sizeof(float);

    // 1) zero the aggregation buffer
    {
        long n4 = (long)Nn * 128 / 4;
        zero_f4_kernel<<<(unsigned)((n4 + 255) / 256), 256, 0, stream>>>((float4*)agg, n4);
    }
    // 2) pack weights into bf16 B fragments
    auto packLaunch = [&](const float* W, __bf16* dst, int K, int N) {
        int threads = (K / 32) * (N / 16) * 32;
        pack_b_kernel<<<(threads + 255) / 256, 256, 0, stream>>>(W, dst, K, N);
    };
    packLaunch(W_lin,  Blin,  128, 128);
    packLaunch(Wt,     Bt,    128, 128);
    packLaunch(W_lin1, Blin1, 128, 128);
    packLaunch(We1,    Be1,   256, 128);
    packLaunch(We2,    Be2,   128, 128);
    packLaunch(Wn1,    Bn1,   256, 128);
    packLaunch(Wn2,    Bn2,   128, 128);
    pack_att_kernel<<<1, 256, 0, stream>>>(W_att, Batt);
    bias_comb_kernel<<<1, 128, 0, stream>>>(b_lin, bt, bc, 128);

    // 3) node projection -> h (fp32 + bf16)
    node_proj_kernel<<<200, 256, 0, stream>>>(x, temb, Blin, Bt, Blin1, bc, h32, hbf, Nn);
    // 4) edge MLP + attention + scatter (persistent, dual-M, weights in LDS)
    edge_kernel<<<640, 192, 0, stream>>>(hbf, row_idx, col_idx, b_att, edge_mask,
                                         Be1, be1, Be2, be2, Batt, agg, Ecnt);
    // 5) node MLP + residual + mask -> out
    node_mlp_kernel<<<200, 256, 0, stream>>>(h32, hbf, agg, Bn1, bn1, Bn2, bn2,
                                             node_mask, (float*)d_out, Nn);
}
